// MultiHeadAttention_34144990003917
// MI455X (gfx1250) — matrix operand order repaired and checked by reading, not compiled, not run on hardware
//
#include <hip/hip_runtime.h>
#include <hip/hip_bf16.h>

#define DM 1024
#define NH 16
#define DH 64
#define BB 4
#define TT 2048
#define BT (BB * TT)

typedef __bf16 bf16;
typedef __attribute__((ext_vector_type(16))) __bf16 bf16x16;
typedef __attribute__((ext_vector_type(8)))  __bf16 bf16x8;
typedef __attribute__((ext_vector_type(4)))  __bf16 bf16x4;
typedef __attribute__((ext_vector_type(8)))  float  f32x8;
typedef __attribute__((ext_vector_type(8)))  int    i32x8;

static __device__ inline f32x8 zero8() {
  f32x8 z = {0.f, 0.f, 0.f, 0.f, 0.f, 0.f, 0.f, 0.f};
  return z;
}

static __device__ inline f32x8 wmma_bf16f32(bf16x16 a, bf16x16 b, f32x8 c) {
  return __builtin_amdgcn_wmma_f32_16x16x32_bf16(false, a, false, b, (short)0, c,
                                                 false, false);
}

static __device__ inline bf16x16 make_frag(bf16x8 lo, bf16x8 hi) {
  bf16x16 f;
#pragma unroll
  for (int i = 0; i < 8; ++i) { f[i] = lo[i]; f[i + 8] = hi[i]; }
  return f;
}

// A fragment (16x32 bf16, row-major [m][k], row stride `stride`):
// lane&15 = M row; lane>=16 picks K slots {8..15, 24..31}.
static __device__ inline bf16x16 fragA(const bf16* p0, int stride, int row0, int k0,
                                       int lane) {
  const bf16* p = p0 + (size_t)(row0 + (lane & 15)) * stride + k0 + ((lane >> 4) << 3);
  return make_frag(*(const bf16x8*)p, *(const bf16x8*)(p + 16));
}

// B fragment (32x16 bf16): lane&15 = N col; lane>=16 picks K slots {8..15, 24..31}. Source staged [k][n].
static __device__ inline bf16x16 fragB(const bf16* p0, int stride, int n0, int k0,
                                       int lane) {
  const bf16* p = p0 + (size_t)(k0 + ((lane >> 4) << 3)) * stride + n0 + (lane & 15);
  bf16x16 f;
#pragma unroll
  for (int i = 0; i < 8; ++i) { f[i] = p[i * stride]; f[i + 8] = p[(i + 16) * stride]; }
  return f;
}

// CDNA5 async copy: 16 bytes per lane, global -> LDS, tracked by ASYNCcnt.
static __device__ inline void async_ld16(const bf16* lds_dst, const bf16* g_src) {
  uint32_t lds_off = (uint32_t)(uintptr_t)lds_dst;  // low 32 bits = LDS offset
  asm volatile("global_load_async_to_lds_b128 %0, %1, off"
               :
               : "v"(lds_off), "v"(g_src)
               : "memory");
}
static __device__ inline void wait_async0() {
  asm volatile("s_wait_asynccnt 0" ::: "memory");
}

// ---------------------------------------------------------------------------
// fp32 -> bf16 elementwise (x4)
// ---------------------------------------------------------------------------
__global__ void cvt_bf16_kernel(const float* __restrict__ in, bf16* __restrict__ out,
                                int n4) {
  int i = blockIdx.x * blockDim.x + threadIdx.x;
  if (i < n4) {
    float4 v = ((const float4*)in)[i];
    bf16x4 o;
    o[0] = (bf16)v.x; o[1] = (bf16)v.y; o[2] = (bf16)v.z; o[3] = (bf16)v.w;
    ((bf16x4*)out)[i] = o;
  }
}

// ---------------------------------------------------------------------------
// fp32 [N][K] -> bf16 [K][N] tiled transpose-convert (weights, one-time)
// block (32,8), tile 32x32 via LDS, coalesced both sides.
// ---------------------------------------------------------------------------
__global__ __launch_bounds__(256) void cvt_transpose_kernel(
    const float* __restrict__ in, bf16* __restrict__ out, int N, int K) {
  __shared__ bf16 tile[32][33];
  const int tx = threadIdx.x, ty = threadIdx.y;
  const int n0 = blockIdx.x * 32, k0 = blockIdx.y * 32;
#pragma unroll
  for (int yy = 0; yy < 4; ++yy)
    tile[ty + yy * 8][tx] = (bf16)in[(size_t)(n0 + ty + yy * 8) * K + k0 + tx];
  __syncthreads();
#pragma unroll
  for (int yy = 0; yy < 4; ++yy)
    out[(size_t)(k0 + ty + yy * 8) * N + n0 + tx] = tile[tx][ty + yy * 8];
}

// ---------------------------------------------------------------------------
// GEMM mainloop (shared by QKV and OUT kernels):
// C[128x128] = X[BT,DM] x Wt[DM,Nw] tile; async staging of both tiles.
// ---------------------------------------------------------------------------
#define GEMM_MAINLOOP(X, Wt, Nw)                                                      \
  constexpr int AS = 72, BS = 136;                                                    \
  __shared__ bf16 sA[128 * AS]; /* [m][k] 128x64 */                                   \
  __shared__ bf16 sB[64 * BS];  /* [k][n] 64x128 */                                   \
  const int tid = threadIdx.x, lane = tid & 31, wid = tid >> 5;                       \
  const int wm = (wid >> 2) * 64, wn = (wid & 3) * 32;                                \
  const int m0 = blockIdx.x * 128, n0 = blockIdx.y * 128;                             \
  f32x8 acc[4][2];                                                                    \
  _Pragma("unroll") for (int i = 0; i < 4; ++i)                                       \
      _Pragma("unroll") for (int j = 0; j < 2; ++j) acc[i][j] = zero8();              \
  for (int k0 = 0; k0 < DM; k0 += 64) {                                               \
    __syncthreads();                                                                  \
    _Pragma("unroll") for (int c = 0; c < 4; ++c) {                                   \
      const int off = (tid + c * 256) * 8;                                            \
      const int row = off >> 6, col = off & 63;                                       \
      async_ld16(&sA[row * AS + col], X + (size_t)(m0 + row) * DM + k0 + col);        \
    }                                                                                 \
    _Pragma("unroll") for (int c = 0; c < 4; ++c) {                                   \
      const int off = (tid + c * 256) * 8;                                            \
      const int row = off >> 7, col = off & 127;                                      \
      async_ld16(&sB[row * BS + col], Wt + (size_t)(k0 + row) * (Nw) + n0 + col);     \
    }                                                                                 \
    wait_async0();                                                                    \
    __syncthreads();                                                                  \
    _Pragma("unroll") for (int kk = 0; kk < 64; kk += 32) {                           \
      bf16x16 af[4], bfm[2];                                                          \
      _Pragma("unroll") for (int i = 0; i < 4; ++i)                                   \
          af[i] = fragA(sA, AS, wm + i * 16, kk, lane);                               \
      _Pragma("unroll") for (int j = 0; j < 2; ++j)                                   \
          bfm[j] = fragB(sB, BS, wn + j * 16, kk, lane);                              \
      _Pragma("unroll") for (int i = 0; i < 4; ++i)                                   \
          _Pragma("unroll") for (int j = 0; j < 2; ++j)                               \
              acc[i][j] = wmma_bf16f32(af[i], bfm[j], acc[i][j]);                     \
    }                                                                                 \
  }

// ---------------------------------------------------------------------------
// QKV GEMM -> scattered bf16 Q/K/V in [B,H,T,dh]; softmax scale folded into Q.
// ---------------------------------------------------------------------------
__global__ __launch_bounds__(256) void qkv_gemm_kernel(
    const bf16* __restrict__ X, const bf16* __restrict__ Wt,
    const float* __restrict__ bias, bf16* __restrict__ Qo, bf16* __restrict__ Ko,
    bf16* __restrict__ Vo) {
  GEMM_MAINLOOP(X, Wt, 3 * DM)
  const int lh = lane >> 4, ln = lane & 15;
#pragma unroll
  for (int j = 0; j < 2; ++j) {
    const int col = n0 + wn + j * 16 + ln;
    const int which = col >> 10;
    const int d = col & 1023;
    const int h = d >> 6, di = d & 63;
    bf16* dst = (which == 0) ? Qo : ((which == 1) ? Ko : Vo);
    const float scale = (which == 0) ? 0.125f : 1.0f;  // 1/sqrt(64)
    const float bv = bias[col];
#pragma unroll
    for (int i = 0; i < 4; ++i) {
#pragma unroll
      for (int r = 0; r < 8; ++r) {
        const int row = m0 + wm + i * 16 + r + lh * 8;
        const int b = row >> 11, t = row & 2047;
        dst[(size_t)((b * NH + h) * TT + t) * DH + di] =
            (bf16)((acc[i][j][r] + bv) * scale);
      }
    }
  }
}

// ---------------------------------------------------------------------------
// Output GEMM: ctx[BT,DM] x Wt[DM,DM] + out_b -> fp32 out [BT,DM]
// ---------------------------------------------------------------------------
__global__ __launch_bounds__(256) void out_gemm_kernel(const bf16* __restrict__ X,
                                                       const bf16* __restrict__ Wt,
                                                       const float* __restrict__ bias,
                                                       float* __restrict__ Out) {
  GEMM_MAINLOOP(X, Wt, DM)
  const int lh = lane >> 4, ln = lane & 15;
#pragma unroll
  for (int j = 0; j < 2; ++j) {
    const int col = n0 + wn + j * 16 + ln;
    const float bv = bias[col];
#pragma unroll
    for (int i = 0; i < 4; ++i) {
#pragma unroll
      for (int r = 0; r < 8; ++r) {
        const int row = m0 + wm + i * 16 + r + lh * 8;
        Out[(size_t)row * DM + col] = acc[i][j][r] + bv;
      }
    }
  }
}

// ---------------------------------------------------------------------------
// Flash attention, transposed-scores formulation. grid=(T/128, B*H), 256 thr.
// Each wave owns a 16-row Q strip; computes S^T = K @ Q^T so K stages row-major
// (async), Q^T staged once per workgroup. Softmax stats are per-lane scalars.
// ---------------------------------------------------------------------------
__global__ __launch_bounds__(256) void attn_kernel(const bf16* __restrict__ Q,
                                                   const bf16* __restrict__ K,
                                                   const bf16* __restrict__ V,
                                                   bf16* __restrict__ Ctx) {
  // layout (bytes): sK [64x72] @0 (9216), sV [64x72] @9216, sP 8x[16x72] @18432
  // pre-loop: Q^T staging (8 waves x 64x24 = 24576 B) overlaps the above.
  constexpr int KS = 72, VS = 72, PS = 72, QS = 24;
  __shared__ bf16 smem[27648];  // 55296 bytes
  bf16* sK = smem;
  bf16* sV = smem + 64 * KS;
  bf16* sP = smem + 2 * 64 * KS;
  const int tid = threadIdx.x, lane = tid & 31, wid = tid >> 5;
  const int lh = lane >> 4, ln = lane & 15;
  const int bh = blockIdx.y;
  const bf16* Qb = Q + (size_t)bh * TT * DH;
  const bf16* Kb = K + (size_t)bh * TT * DH;
  const bf16* Vb = V + (size_t)bh * TT * DH;
  const int q0 = blockIdx.x * 128 + wid * 16;

  // ---- one-time: stage this wave's Q strip transposed, grab B-fragments ----
  bf16x16 qtf0, qtf1;
  {
    bf16* qt = smem + wid * (64 * QS);  // [k=dh][m=q] stride QS
    const bf16* qrow = Qb + (size_t)(q0 + ln) * DH + lh * 32;
    bf16x8 c0 = *(const bf16x8*)qrow;
    bf16x8 c1 = *(const bf16x8*)(qrow + 8);
    bf16x8 c2 = *(const bf16x8*)(qrow + 16);
    bf16x8 c3 = *(const bf16x8*)(qrow + 24);
#pragma unroll
    for (int e = 0; e < 8; ++e) {
      qt[(lh * 32 + e)      * QS + ln] = c0[e];
      qt[(lh * 32 + e + 8)  * QS + ln] = c1[e];
      qt[(lh * 32 + e + 16) * QS + ln] = c2[e];
      qt[(lh * 32 + e + 24) * QS + ln] = c3[e];
    }
    __syncthreads();
    qtf0 = fragB(qt, QS, 0, 0, lane);
    qtf1 = fragB(qt, QS, 0, 32, lane);
    // force materialization so the ds_loads complete before smem is reused
    i32x8 t0 = __builtin_bit_cast(i32x8, qtf0);
    i32x8 t1 = __builtin_bit_cast(i32x8, qtf1);
    asm volatile("" : : "v"(t0), "v"(t1));
  }

  f32x8 st[4], o[4];
#pragma unroll
  for (int u = 0; u < 4; ++u) o[u] = zero8();
  float m_i = -3.0e38f, l_i = 0.f;  // per-lane: q row = ln
  bf16* myP = sP + wid * 16 * PS;   // [m=q 16][k=key 64]

  for (int kt = 0; kt < TT; kt += 64) {
    __syncthreads();  // all waves done reading sK/sV (or q^T pre-loop)
#pragma unroll
    for (int c = 0; c < 2; ++c) {  // async-stage K and V tiles (64x64 each)
      const int off = (tid + c * 256) * 8;
      const int row = off >> 6, col = off & 63;
      async_ld16(&sK[row * KS + col], Kb + (size_t)(kt + row) * DH + col);
      async_ld16(&sV[row * VS + col], Vb + (size_t)(kt + row) * DH + col);
    }
    wait_async0();
    __syncthreads();

    // S^T[key][q] = K @ Q^T (64x16), Q carries 1/sqrt(dh)
#pragma unroll
    for (int j = 0; j < 4; ++j) {
      st[j] = wmma_bf16f32(fragA(sK, KS, j * 16, 0, lane), qtf0, zero8());
      st[j] = wmma_bf16f32(fragA(sK, KS, j * 16, 32, lane), qtf1, st[j]);
    }

    // online softmax: this lane holds 32 keys of row q=ln; other 32 in lane^16
    float mx = st[0][0];
#pragma unroll
    for (int j = 0; j < 4; ++j)
#pragma unroll
      for (int r = 0; r < 8; ++r) mx = fmaxf(mx, st[j][r]);
    mx = fmaxf(mx, __shfl_xor(mx, 16, 32));
    const float mnew = fmaxf(m_i, mx);
    const float alpha = __expf(m_i - mnew);
    m_i = mnew;
    float rs = 0.f;
#pragma unroll
    for (int j = 0; j < 4; ++j) {
      bf16x8 pv;
#pragma unroll
      for (int r = 0; r < 8; ++r) {
        const float p = __expf(st[j][r] - mnew);
        st[j][r] = p;
        rs += p;
        pv[r] = (bf16)p;
      }
      // 8 register rows of S^T tile j = 8 consecutive key cols of P row ln
      *(bf16x8*)&myP[ln * PS + j * 16 + lh * 8] = pv;
    }
    rs += __shfl_xor(rs, 16, 32);
    l_i = l_i * alpha + rs;

    // rescale O: alpha lives in lane-space (q=ln), O rows live in reg-space
    float ar[8];
#pragma unroll
    for (int r = 0; r < 8; ++r) ar[r] = __shfl(alpha, r + lh * 8, 32);
#pragma unroll
    for (int u = 0; u < 4; ++u)
#pragma unroll
      for (int r = 0; r < 8; ++r) o[u][r] *= ar[r];

    // O += P @ V  (16x64)
#pragma unroll
    for (int u = 0; u < 4; ++u)
#pragma unroll
      for (int kk = 0; kk < 2; ++kk)
        o[u] = wmma_bf16f32(fragA(myP, PS, 0, kk * 32, lane),
                            fragB(sV, VS, u * 16, kk * 32, lane), o[u]);
  }

  // normalize rows (l in lane-space -> reg-space) and store ctx bf16 [B,T,D]
  float linv[8];
#pragma unroll
  for (int r = 0; r < 8; ++r) linv[r] = 1.0f / __shfl(l_i, r + lh * 8, 32);
  const int h = bh & (NH - 1), b = bh >> 4;
#pragma unroll
  for (int u = 0; u < 4; ++u)
#pragma unroll
    for (int r = 0; r < 8; ++r) {
      const int row = q0 + r + lh * 8;
      Ctx[((size_t)b * TT + row) * DM + h * DH + u * 16 + ln] =
          (bf16)(o[u][r] * linv[r]);
    }
}

// ---------------------------------------------------------------------------
extern "C" void kernel_launch(void* const* d_in, const int* in_sizes, int n_in,
                              void* d_out, int out_size, void* d_ws, size_t ws_size,
                              hipStream_t stream) {
  (void)in_sizes; (void)n_in; (void)out_size; (void)ws_size;
  const float* x     = (const float*)d_in[0];
  const float* qkv_w = (const float*)d_in[1];
  const float* qkv_b = (const float*)d_in[2];
  const float* out_w = (const float*)d_in[3];
  const float* out_b = (const float*)d_in[4];
  float* out = (float*)d_out;

  bf16* xb    = (bf16*)d_ws;                  // BT*DM
  bf16* wqkvt = xb + (size_t)BT * DM;         // [DM][3DM] transposed
  bf16* wot   = wqkvt + (size_t)3 * DM * DM;  // [DM][DM] transposed
  bf16* Qb    = wot + (size_t)DM * DM;        // BT*DM each, [B,H,T,dh]
  bf16* Kb    = Qb + (size_t)BT * DM;
  bf16* Vb    = Kb + (size_t)BT * DM;
  bf16* ctxb  = Vb + (size_t)BT * DM;

  cvt_bf16_kernel<<<(BT * DM / 4 + 255) / 256, 256, 0, stream>>>(x, xb, BT * DM / 4);
  cvt_transpose_kernel<<<dim3(3 * DM / 32, DM / 32), dim3(32, 8), 0, stream>>>(
      qkv_w, wqkvt, 3 * DM, DM);
  cvt_transpose_kernel<<<dim3(DM / 32, DM / 32), dim3(32, 8), 0, stream>>>(
      out_w, wot, DM, DM);

  qkv_gemm_kernel<<<dim3(BT / 128, 3 * DM / 128), 256, 0, stream>>>(
      xb, wqkvt, qkv_b, Qb, Kb, Vb);
  attn_kernel<<<dim3(TT / 128, BB * NH), 256, 0, stream>>>(Qb, Kb, Vb, ctxb);
  out_gemm_kernel<<<dim3(BT / 128, DM / 128), 256, 0, stream>>>(ctxb, wot, out_b, out);
}
